// VCTreePredictor_16269336118079
// MI455X (gfx1250) — matrix operand order repaired and checked by reading, not compiled, not run on hardware
//
#include <hip/hip_runtime.h>
#include <hip/hip_bf16.h>
#include <stdint.h>

// Problem dims
#define HH    512          // H
#define H2    1024         // 2*H
#define PDIM  4096         // P
#define NREL  51
#define NRELP 64           // padded NUM_REL
#define NOBJ  151
#define NBAT  512          // B
#define NPER  64
#define RPER  128
#define NTOT  (NBAT*NPER)  // 32768 objects
#define RTOT  (NBAT*RPER)  // 65536 pairs

typedef __attribute__((ext_vector_type(8)))  float  v8f;
typedef __attribute__((ext_vector_type(16))) __bf16 v16bf;
typedef __attribute__((ext_vector_type(4)))  unsigned int u32x4;
typedef __attribute__((ext_vector_type(8)))  int i32x8;
typedef __attribute__((ext_vector_type(4)))  int i32x4;

union V16 {
  v16bf v;
  unsigned short s[16];
  uint4 q[2];
};

__device__ __forceinline__ unsigned short f2bf(float f) {
  unsigned u = __builtin_bit_cast(unsigned, f);
  u += 0x7fffu + ((u >> 16) & 1u);     // round-to-nearest-even
  return (unsigned short)(u >> 16);
}
__device__ __forceinline__ float bf2f(unsigned short s) {
  unsigned u = ((unsigned)s) << 16;
  return __builtin_bit_cast(float, u);
}

__device__ __forceinline__ v8f vzero() {
  v8f z;
#pragma unroll
  for (int i = 0; i < 8; ++i) z[i] = 0.0f;
  return z;
}

__device__ __forceinline__ v8f wmma_bf16(v16bf a, v16bf b, v8f c) {
  // D = A(16x32 bf16) * B(32x16 bf16) + C(f32)
  return __builtin_amdgcn_wmma_f32_16x16x32_bf16(
      false, a, false, b, (short)0, c, false, false);
}

// Split two v8f (A-frag halves) into bf16 hi/lo fragments (3xBF16 emulation).
__device__ __forceinline__ void split_pair(v8f x0, v8f x1, V16& hi, V16& lo) {
#pragma unroll
  for (int i = 0; i < 8; ++i) {
    unsigned short hb = f2bf(x0[i]);
    hi.s[i] = hb;
    lo.s[i] = f2bf(x0[i] - bf2f(hb));
  }
#pragma unroll
  for (int i = 0; i < 8; ++i) {
    unsigned short hb = f2bf(x1[i]);
    hi.s[8 + i] = hb;
    lo.s[8 + i] = f2bf(x1[i] - bf2f(hb));
  }
}

// ---------------------------------------------------------------------------
// TDM: DMA a 2-D tile (tile_y rows x tile_x bf16 elements) of a row-major
// [tensor_rows][tensor_cols] bf16 tensor into LDS (packed row-major).
// Descriptor per CDNA5 ISA. Issued by one wave,
// completion tracked with TENSORcnt. 6-arg builtin form (clang-23 /
// therock-10.0 headers): (g0, g1, g2, g3, g4, cpol).
// ---------------------------------------------------------------------------
__device__ __forceinline__ void tdm_load_tile(const unsigned short* gbase,
                                              unsigned short* lds_dst,
                                              int row0, int col0,
                                              int tensor_cols, int tensor_rows,
                                              int tile_x, int tile_y) {
  unsigned long long ga = (unsigned long long)gbase +
      ((unsigned long long)row0 * (unsigned)tensor_cols + (unsigned)col0) * 2ull;
  unsigned lds = (unsigned)(size_t)lds_dst;  // flat->LDS: low 32 bits
  u32x4 g0;
  g0[0] = 1u;                                  // count=1, user descriptor
  g0[1] = lds;                                 // lds_addr [63:32]
  g0[2] = (unsigned)(ga & 0xffffffffull);      // global_addr[31:0]
  g0[3] = (unsigned)((ga >> 32) & 0x01ffffffull) | (2u << 30);  // type=2
  i32x8 g1;
  g1[0] = (int)(1u << 16);                     // data_size=1 -> 2 bytes
  g1[1] = (int)(((unsigned)tensor_cols & 0xffffu) << 16);       // dim0[15:0]
  g1[2] = (int)((((unsigned)tensor_cols >> 16) & 0xffffu) |
                (((unsigned)tensor_rows & 0xffffu) << 16));     // dim0 hi|dim1 lo
  g1[3] = (int)(((((unsigned)tensor_rows) >> 16) & 0xffffu) |
                (((unsigned)tile_x & 0xffffu) << 16));          // dim1 hi|tile0
  g1[4] = (int)((unsigned)tile_y & 0xffffu);   // tile_dim1 (tile_dim2=0)
  g1[5] = tensor_cols;                         // dim0_stride[31:0]
  g1[6] = 0;                                   // dim0_stride hi | dim1_stride lo
  g1[7] = 0;
  i32x4 z4;
  z4[0] = 0; z4[1] = 0; z4[2] = 0; z4[3] = 0;
  i32x8 z8;
#pragma unroll
  for (int i = 0; i < 8; ++i) z8[i] = 0;
  __builtin_amdgcn_tensor_load_to_lds(g0, g1, z4, z4, z8, 0);
}

// ---------------------------------------------------------------------------
// Weight prep: W[K][Nsrc] (row-major fp32) -> W^T hi/lo bf16 [Npad][K]
// ---------------------------------------------------------------------------
__global__ void split_tr_kernel(const float* __restrict__ W,
                                unsigned short* __restrict__ Thi,
                                unsigned short* __restrict__ Tlo,
                                int K, int Nsrc, int Npad) {
  long long idx = (long long)blockIdx.x * blockDim.x + threadIdx.x;
  long long total = (long long)Npad * K;
  if (idx >= total) return;
  int n = (int)(idx / K);
  int k = (int)(idx % K);
  float x = (n < Nsrc) ? W[(long long)k * Nsrc + n] : 0.0f;
  unsigned short hb = f2bf(x);
  Thi[idx] = hb;
  Tlo[idx] = f2bf(x - bf2f(hb));
}

// ---------------------------------------------------------------------------
// GEMM1: edge_rep = relu(edge_ctx[32768,512] @ W_pe[512,1024] + b_pe)
// Output written pre-split as bf16 hi/lo so the fused kernel's A-fragments
// need zero conversion VALU.
// ---------------------------------------------------------------------------
__global__ __launch_bounds__(256) void gemm_pe_kernel(
    const float* __restrict__ edge_ctx, const float* __restrict__ b_pe,
    const unsigned short* __restrict__ WT_hi,
    const unsigned short* __restrict__ WT_lo,
    unsigned short* __restrict__ edge_hi, unsigned short* __restrict__ edge_lo) {
  __shared__ unsigned short Bs_hi[128 * 32];
  __shared__ unsigned short Bs_lo[128 * 32];

  const int t = threadIdx.x;
  const int lane = t & 31, wave = t >> 5;
  const int lam = lane & 15, hh = lane >> 4;
  const int mb = blockIdx.x * 128, nbv = blockIdx.y * 128;
  const int mrow = mb + wave * 16 + lam;
  const int srow = t >> 1, shalf = t & 1;

  v8f acc[8];
#pragma unroll
  for (int nn = 0; nn < 8; ++nn) acc[nn] = vzero();

  for (int kc = 0; kc < 16; ++kc) {
    const int k0 = kc * 32;
    __syncthreads();
    {  // stage W_pe^T slice [nbv..+127][k0..+31] hi+lo
      size_t s = (size_t)(nbv + srow) * HH + (size_t)(k0 + shalf * 16);
      int d = srow * 32 + shalf * 16;
      ((uint4*)(Bs_hi + d))[0] = ((const uint4*)(WT_hi + s))[0];
      ((uint4*)(Bs_hi + d))[1] = ((const uint4*)(WT_hi + s))[1];
      ((uint4*)(Bs_lo + d))[0] = ((const uint4*)(WT_lo + s))[0];
      ((uint4*)(Bs_lo + d))[1] = ((const uint4*)(WT_lo + s))[1];
    }
    __syncthreads();
    const float* ap = edge_ctx + (size_t)mrow * HH + k0 + hh * 16;
    v8f a0 = *((const v8f*)ap);
    v8f a1 = *((const v8f*)(ap + 8));
    V16 ahi, alo;
    split_pair(a0, a1, ahi, alo);
#pragma unroll
    for (int nn = 0; nn < 8; ++nn) {
      V16 bh, bl;
      const uint4* ph = (const uint4*)(Bs_hi + (nn * 16 + lam) * 32 + hh * 16);
      bh.q[0] = ph[0]; bh.q[1] = ph[1];
      const uint4* pl = (const uint4*)(Bs_lo + (nn * 16 + lam) * 32 + hh * 16);
      bl.q[0] = pl[0]; bl.q[1] = pl[1];
      acc[nn] = wmma_bf16(ahi.v, bh.v, acc[nn]);
      acc[nn] = wmma_bf16(ahi.v, bl.v, acc[nn]);
      acc[nn] = wmma_bf16(alo.v, bh.v, acc[nn]);
    }
  }

#pragma unroll
  for (int nn = 0; nn < 8; ++nn) {
    int col = nbv + nn * 16 + lam;
    float bias = b_pe[col];
#pragma unroll
    for (int g = 0; g < 8; ++g) {
      int row = mb + wave * 16 + g + hh * 8;
      float vv = acc[nn][g] + bias;
      vv = vv > 0.0f ? vv : 0.0f;
      unsigned short hb = f2bf(vv);
      size_t o = (size_t)row * H2 + col;
      edge_hi[o] = hb;
      edge_lo[o] = f2bf(vv - bf2f(hb));
    }
  }
}

// ---------------------------------------------------------------------------
// Fused: gather -> (A @ W_pc + b_pc) * union -> @ W_cc + b_cc + freq
// One block per image (128 pairs), 8 waves. W_pc tiles double-buffered into
// LDS by the Tensor Data Mover (wave 0 issues, TENSORcnt-synced).
// ---------------------------------------------------------------------------
__global__ __launch_bounds__(256) void fused_rel_kernel(
    const unsigned short* __restrict__ edge_hi,
    const unsigned short* __restrict__ edge_lo,
    const float* __restrict__ union_f,
    const unsigned short* __restrict__ WpcT_hi,
    const unsigned short* __restrict__ WpcT_lo,
    const float* __restrict__ b_pc,
    const unsigned short* __restrict__ WccT_hi,
    const unsigned short* __restrict__ WccT_lo,
    const float* __restrict__ b_cc, const float* __restrict__ freq_table,
    const int* __restrict__ obj_preds, const int* __restrict__ rel_pairs,
    float* __restrict__ out) {
  __shared__ unsigned short Bs_hi[2][128 * 32];  // 16 KB (double-buffered)
  __shared__ unsigned short Bs_lo[2][128 * 32];  // 16 KB
  __shared__ unsigned Pbuf[8 * 16 * 32];         // 16 KB (per-wave 16x32)
  __shared__ int hIdx[RPER], tIdx[RPER], plab[RPER];

  const int t = threadIdx.x;
  const int lane = t & 31, wave = t >> 5;
  const int lam = lane & 15, hh = lane >> 4;
  const int blk = blockIdx.x;
  const int rbase = blk * RPER;

  if (t < RPER) {
    int r = rbase + t;
    int off = blk * NPER;
    int hv = rel_pairs[2 * r] + off;
    int tv = rel_pairs[2 * r + 1] + off;
    hIdx[t] = hv;
    tIdx[t] = tv;
    plab[t] = obj_preds[hv] * NOBJ + obj_preds[tv];
  }
  __syncthreads();

  v8f cacc[4];  // ctx accumulators: 16 rows x 64 cols (51 used)
#pragma unroll
  for (int j = 0; j < 4; ++j) cacc[j] = vzero();

  const int mrl = wave * 16 + lam;
  unsigned* Pw = Pbuf + wave * (16 * 32);

  for (int nt = 0; nt < 32; ++nt) {  // N-tiles of 128 over P=4096
    const int nbase = nt * 128;
    v8f acc[8];
#pragma unroll
    for (int nn = 0; nn < 8; ++nn) acc[nn] = vzero();

    // Preload K-chunk 0 via TDM (buffers were drained >=1 barrier ago)
    if (wave == 0) {
      tdm_load_tile(WpcT_hi, &Bs_hi[0][0], nbase, 0, H2, PDIM, 32, 128);
      tdm_load_tile(WpcT_lo, &Bs_lo[0][0], nbase, 0, H2, PDIM, 32, 128);
    }

    for (int kc = 0; kc < 32; ++kc) {  // K chunks of 32 over 2H=1024
      const int k0 = kc * 32;
      const int buf = kc & 1;
      if (wave == 0) __builtin_amdgcn_s_wait_tensorcnt(0);
      __syncthreads();  // chunk kc ready for all waves; kc-1 buffer drained
      if (wave == 0 && kc + 1 < 32) {  // prefetch next chunk into other buffer
        tdm_load_tile(WpcT_hi, &Bs_hi[buf ^ 1][0], nbase, k0 + 32, H2, PDIM, 32, 128);
        tdm_load_tile(WpcT_lo, &Bs_lo[buf ^ 1][0], nbase, k0 + 32, H2, PDIM, 32, 128);
      }
      // gathered A fragments, pre-split bf16: zero conversion VALU
      int ridx = (k0 < HH) ? hIdx[mrl] : tIdx[mrl];
      size_t abase = (size_t)ridx * H2 + k0 + hh * 16;
      V16 ahi, alo;
      ahi.q[0] = *((const uint4*)(edge_hi + abase));
      ahi.q[1] = *((const uint4*)(edge_hi + abase + 8));
      alo.q[0] = *((const uint4*)(edge_lo + abase));
      alo.q[1] = *((const uint4*)(edge_lo + abase + 8));
#pragma unroll
      for (int nn = 0; nn < 8; ++nn) {
        V16 bh, bl;
        const uint4* ph =
            (const uint4*)(&Bs_hi[buf][0] + (nn * 16 + lam) * 32 + hh * 16);
        bh.q[0] = ph[0]; bh.q[1] = ph[1];
        const uint4* pl =
            (const uint4*)(&Bs_lo[buf][0] + (nn * 16 + lam) * 32 + hh * 16);
        bl.q[0] = pl[0]; bl.q[1] = pl[1];
        acc[nn] = wmma_bf16(ahi.v, bh.v, acc[nn]);
        acc[nn] = wmma_bf16(ahi.v, bl.v, acc[nn]);
        acc[nn] = wmma_bf16(alo.v, bh.v, acc[nn]);
      }
    }

    // Epilogue fused with GEMM3, 32 columns (2 subtiles) at a time:
    //   prod = (acc + b_pc) * union  -> packed bf16 hi|lo in per-wave LDS
    //   cacc += prod(16x32) @ W_cc^T chunk   (split bf16, 3 WMMAs/subtile)
#pragma unroll
    for (int q = 0; q < 4; ++q) {
#pragma unroll
      for (int nn = 0; nn < 2; ++nn) {
        int nn8 = q * 2 + nn;
        int col = nbase + nn8 * 16 + lam;
        float bias = b_pc[col];
#pragma unroll
        for (int g = 0; g < 8; ++g) {
          int rl = wave * 16 + g + hh * 8;
          int row = rbase + rl;
          float u = union_f[(size_t)row * PDIM + col];
          if (nt < 31)  // prefetch next union tile (global_prefetch_b8)
            __builtin_prefetch(&union_f[(size_t)row * PDIM + col + 128], 0, 3);
          float p = (acc[nn8][g] + bias) * u;
          unsigned short phb = f2bf(p);
          unsigned short plb = f2bf(p - bf2f(phb));
          Pw[(g + hh * 8) * 32 + nn * 16 + lam] =
              (unsigned)phb | ((unsigned)plb << 16);
        }
      }
      // GEMM3 on this 32-col chunk (same-wave LDS RAW: in-order)
      const unsigned* pr = Pw + lam * 32 + hh * 16;
      V16 phi, plo;
#pragma unroll
      for (int i = 0; i < 8; ++i) {
        unsigned w0 = pr[i];
        phi.s[i] = (unsigned short)(w0 & 0xffffu);
        plo.s[i] = (unsigned short)(w0 >> 16);
      }
#pragma unroll
      for (int i = 0; i < 8; ++i) {
        unsigned w0 = pr[8 + i];
        phi.s[8 + i] = (unsigned short)(w0 & 0xffffu);
        plo.s[8 + i] = (unsigned short)(w0 >> 16);
      }
      int kg = nbase + q * 32;
#pragma unroll
      for (int jn = 0; jn < 4; ++jn) {
        V16 bh, bl;
        const uint4* ph =
            (const uint4*)(WccT_hi + (size_t)(jn * 16 + lam) * PDIM + kg + hh * 16);
        bh.q[0] = ph[0]; bh.q[1] = ph[1];
        const uint4* pl =
            (const uint4*)(WccT_lo + (size_t)(jn * 16 + lam) * PDIM + kg + hh * 16);
        bl.q[0] = pl[0]; bl.q[1] = pl[1];
        cacc[jn] = wmma_bf16(phi.v, bh.v, cacc[jn]);
        cacc[jn] = wmma_bf16(phi.v, bl.v, cacc[jn]);
        cacc[jn] = wmma_bf16(plo.v, bh.v, cacc[jn]);
      }
    }
  }

  // Final epilogue: + b_cc + freq_table[pair_lab], write 51 cols
#pragma unroll
  for (int jn = 0; jn < 4; ++jn) {
    int col = jn * 16 + lam;
    if (col < NREL) {
      float bias = b_cc[col];
#pragma unroll
      for (int g = 0; g < 8; ++g) {
        int rl = wave * 16 + g + hh * 8;
        int row = rbase + rl;
        out[(size_t)row * NREL + col] =
            cacc[jn][g] + bias + freq_table[(size_t)plab[rl] * NREL + col];
      }
    }
  }
}

// ---------------------------------------------------------------------------
extern "C" void kernel_launch(void* const* d_in, const int* in_sizes, int n_in,
                              void* d_out, int out_size, void* d_ws,
                              size_t ws_size, hipStream_t stream) {
  (void)in_sizes; (void)n_in; (void)out_size; (void)ws_size;
  const float* edge_ctx = (const float*)d_in[0];
  const float* union_f  = (const float*)d_in[1];
  const float* W_pe     = (const float*)d_in[2];
  const float* b_pe     = (const float*)d_in[3];
  const float* W_pc     = (const float*)d_in[4];
  const float* b_pc     = (const float*)d_in[5];
  const float* W_cc     = (const float*)d_in[6];
  const float* b_cc     = (const float*)d_in[7];
  const float* freq     = (const float*)d_in[8];
  const int*   obj_preds = (const int*)d_in[9];
  const int*   rel_pairs = (const int*)d_in[10];
  float* out = (float*)d_out;

  // Workspace layout (~147 MB)
  char* ws = (char*)d_ws;
  size_t off = 0;
  unsigned short* edge_hi = (unsigned short*)(ws + off); off += (size_t)NTOT * H2 * 2;
  unsigned short* edge_lo = (unsigned short*)(ws + off); off += (size_t)NTOT * H2 * 2;
  unsigned short* WpeT_hi = (unsigned short*)(ws + off); off += (size_t)H2 * HH * 2;
  unsigned short* WpeT_lo = (unsigned short*)(ws + off); off += (size_t)H2 * HH * 2;
  unsigned short* WpcT_hi = (unsigned short*)(ws + off); off += (size_t)PDIM * H2 * 2;
  unsigned short* WpcT_lo = (unsigned short*)(ws + off); off += (size_t)PDIM * H2 * 2;
  unsigned short* WccT_hi = (unsigned short*)(ws + off); off += (size_t)NRELP * PDIM * 2;
  unsigned short* WccT_lo = (unsigned short*)(ws + off); off += (size_t)NRELP * PDIM * 2;

  // 1) split+transpose weights to bf16 hi/lo (one-shot, deterministic)
  {
    long long tot = (long long)H2 * HH;  // W_pe^T [1024][512]
    split_tr_kernel<<<(unsigned)((tot + 255) / 256), 256, 0, stream>>>(
        W_pe, WpeT_hi, WpeT_lo, HH, H2, H2);
  }
  {
    long long tot = (long long)PDIM * H2;  // W_pc^T [4096][1024]
    split_tr_kernel<<<(unsigned)((tot + 255) / 256), 256, 0, stream>>>(
        W_pc, WpcT_hi, WpcT_lo, H2, PDIM, PDIM);
  }
  {
    long long tot = (long long)NRELP * PDIM;  // W_cc^T [64][4096], zero-padded
    split_tr_kernel<<<(unsigned)((tot + 255) / 256), 256, 0, stream>>>(
        W_cc, WccT_hi, WccT_lo, PDIM, NREL, NRELP);
  }

  // 2) edge_rep = relu(edge_ctx @ W_pe + b_pe), stored pre-split bf16 hi/lo
  dim3 g1(NTOT / 128, H2 / 128);  // (256, 8)
  gemm_pe_kernel<<<g1, 256, 0, stream>>>(edge_ctx, b_pe, WpeT_hi, WpeT_lo,
                                         edge_hi, edge_lo);

  // 3) fused gather + GEMM2 (TDM-staged B) + union-mul + GEMM3 + freq add
  fused_rel_kernel<<<RTOT / RPER, 256, 0, stream>>>(
      edge_hi, edge_lo, union_f, WpcT_hi, WpcT_lo, b_pc, WccT_hi, WccT_lo,
      b_cc, freq, obj_preds, rel_pairs, out);
}
